// GroupQueryAttention_73796128080177
// MI455X (gfx1250) — matrix operand order repaired and checked by reading, not compiled, not run on hardware
//
#include <hip/hip_runtime.h>

// ---------------------------------------------------------------------------
// GQA for MI455X (gfx1250): bf16 WMMA everywhere, fp32 accumulate.
//   S=2048, HIDDEN=4096, H=32 heads, G=8 groups, D=128, rep=4.
// Compute-bound (~240 GFLOP vs ~200MB mandatory HBM traffic), so: convert
// fp32->bf16 while staging to LDS, double-buffer all staging, and use the
// Tensor Data Mover (async, TENSORcnt) for the 2-D K/Q tiles in attention.
// ---------------------------------------------------------------------------

typedef __attribute__((ext_vector_type(16))) __bf16 v16bf;
typedef __attribute__((ext_vector_type(8)))  __bf16 v8bf;
typedef __attribute__((ext_vector_type(8)))  float  v8f;

union ABf { v16bf v; v8bf h[2]; };

__device__ __forceinline__ v8f wmma_bf16(const ABf& a, const ABf& b, v8f c) {
  return __builtin_amdgcn_wmma_f32_16x16x32_bf16(
      /*neg_a=*/false, a.v, /*neg_b=*/false, b.v,
      /*c_mod=*/(short)0, c, /*reuse_a=*/false, /*reuse_b=*/false);
}

__device__ __forceinline__ v8bf cvt8(float4 a, float4 b) {
  v8bf r;
  r[0] = (__bf16)a.x; r[1] = (__bf16)a.y; r[2] = (__bf16)a.z; r[3] = (__bf16)a.w;
  r[4] = (__bf16)b.x; r[5] = (__bf16)b.y; r[6] = (__bf16)b.z; r[7] = (__bf16)b.w;
  return r;
}

// ---------------- Tensor Data Mover (gfx1250) ------------------------------
#if defined(__has_builtin)
# if __has_builtin(__builtin_amdgcn_tensor_load_to_lds) && \
     __has_builtin(__builtin_amdgcn_s_wait_tensorcnt)
#  define GQA_TDM 1
# endif
#endif
#ifndef GQA_TDM
# define GQA_TDM 0
#endif

#if GQA_TDM
typedef unsigned int tdm_v4ui __attribute__((ext_vector_type(4)));
typedef int          tdm_v8i  __attribute__((ext_vector_type(8)));
typedef int          tdm_v4i  __attribute__((ext_vector_type(4)));

__device__ __forceinline__ unsigned lds_byte_off(const void* p) {
  // generic LDS address: low 32 bits are the LDS byte offset
  return (unsigned)(unsigned long long)p;
}

// 2-D TDM load: 128 x 128 bf16 tile, row stride 128 elems in memory,
// LDS destination padded +16B per 256B row (matches DPAD=136 layout).
__device__ __forceinline__ void tdm_load_tile128(const __bf16* g, unsigned lds_off,
                                                 int tensor_rows) {
  unsigned long long ga = (unsigned long long)g;
  tdm_v4ui g0;
  g0[0] = 1u;                                   // count=1 (valid), user mode
  g0[1] = lds_off;                              // lds_addr (bytes)
  g0[2] = (unsigned)ga;                         // global_addr[31:0]  -> bits 95:64
  g0[3] = (unsigned)((ga >> 32) & 0x01FFFFFFull) | (2u << 30); // ga[56:32] | type=2
  tdm_v8i g1;
  // data_size=1(2B) | pad_enable | pad_interval=5(64 DW=256B) | pad_amount=3(4 DW=16B)
  g1[0] = (int)((1u << 16) | (1u << 20) | (5u << 22) | (3u << 25));
  g1[1] = (int)(128u << 16);                    // tensor_dim0=128 (low 16 @ bits 63:48)
  g1[2] = (int)((unsigned)tensor_rows << 16);   // tensor_dim0 hi=0 | tensor_dim1 lo16
  g1[3] = (int)((((unsigned)tensor_rows >> 16) & 0xFFFFu) | (128u << 16)); // td1 hi | tile_dim0=128
  g1[4] = 128;                                  // tile_dim1=128 | tile_dim2=0
  g1[5] = 128;                                  // tensor_dim0_stride low32 = 128
  g1[6] = 0;                                    // td0s hi | td1s lo
  g1[7] = 0;
  tdm_v4i z = {0, 0, 0, 0};
#if defined(__clang_major__) && (__clang_major__ >= 23)
  tdm_v8i z8 = {0, 0, 0, 0, 0, 0, 0, 0};
  __builtin_amdgcn_tensor_load_to_lds(g0, g1, z, z, z8, 0);
#else
  __builtin_amdgcn_tensor_load_to_lds(g0, g1, z, z, 0);
#endif
}
#endif  // GQA_TDM

// ---------------------------------------------------------------------------
// GEMM: Y[M,N] = X[M,K] * W[N,K]^T + bias[N]
//  A_BF16=0: X fp32 (converted to bf16 at LDS store);  =1: X already bf16.
//  OUT_F32=0: store bf16 head-major  out[(n>>7)*M + m][n&127]
//  OUT_F32=1: store fp32 row-major   out[m*N + n]
// Block tile 128x128, K-step 32; 8 waves, each 32(M)x64(N); LDS ping-pong so
// the k+32 staging overlaps the k compute (one barrier per K-step).
// ---------------------------------------------------------------------------
#define LDSTR 40   // 32 bf16 + 8 pad = 80 B rows (16B aligned, conflict-free)

template<bool A_BF16, bool OUT_F32>
__global__ __launch_bounds__(256)
void gqa_gemm_xwt(const void* __restrict__ Ain, const float* __restrict__ W,
                  const float* __restrict__ bias, void* __restrict__ out,
                  int M, int N, int K) {
  __shared__ __bf16 lA[2][128 * LDSTR];
  __shared__ __bf16 lB[2][128 * LDSTR];

  const int tid   = threadIdx.x;
  const int lane  = tid & 31;
  const int wv    = tid >> 5;        // 0..7
  const int wm    = wv >> 1;         // 0..3  (M subtile)
  const int wn    = wv & 1;          // 0..1  (N subtile)
  const int halfl = lane & 15;
  const int hsel  = lane >> 4;       // 0 or 1
  const int m0    = blockIdx.y * 128;
  const int n0    = blockIdx.x * 128;

  const int ldrow = tid >> 1;        // 0..127  (loader row)
  const int ldseg = tid & 1;         // 0..1    (16-elem segment)

  const float*  Af = (const float*)Ain;
  const __bf16* Ab = (const __bf16*)Ain;

  auto stage = [&](int buf, int k0) {
    if constexpr (A_BF16) {
      const v8bf* src = (const v8bf*)(Ab + (size_t)(m0 + ldrow) * K + k0 + ldseg * 16);
      *(v8bf*)&lA[buf][ldrow * LDSTR + ldseg * 16]     = src[0];
      *(v8bf*)&lA[buf][ldrow * LDSTR + ldseg * 16 + 8] = src[1];
    } else {
      const float4* src = (const float4*)(Af + (size_t)(m0 + ldrow) * K + k0 + ldseg * 16);
      float4 f0 = src[0], f1 = src[1], f2 = src[2], f3 = src[3];
      *(v8bf*)&lA[buf][ldrow * LDSTR + ldseg * 16]     = cvt8(f0, f1);
      *(v8bf*)&lA[buf][ldrow * LDSTR + ldseg * 16 + 8] = cvt8(f2, f3);
    }
    const float4* srcw = (const float4*)(W + (size_t)(n0 + ldrow) * K + k0 + ldseg * 16);
    float4 w0 = srcw[0], w1 = srcw[1], w2 = srcw[2], w3 = srcw[3];
    *(v8bf*)&lB[buf][ldrow * LDSTR + ldseg * 16]     = cvt8(w0, w1);
    *(v8bf*)&lB[buf][ldrow * LDSTR + ldseg * 16 + 8] = cvt8(w2, w3);
    if (k0 + 32 < K)  // gfx1250 global_prefetch_b8 of the K-step after next
      __builtin_prefetch(W + (size_t)(n0 + ldrow) * K + k0 + 32 + ldseg * 16, 0, 0);
  };

  v8f acc[2][4] = {};

  stage(0, 0);
  __syncthreads();
  int cur = 0;

  for (int k0 = 0; k0 < K; k0 += 32) {
    if (k0 + 32 < K) stage(cur ^ 1, k0 + 32);   // overlap next-step staging

    // ---- A fragments: 16x32 bf16, lane-half picks K {0..7,16..23}/{8..15,24..31}
    ABf afr[2];
#pragma unroll
    for (int mt = 0; mt < 2; ++mt) {
      const int row  = wm * 32 + mt * 16 + halfl;
      const int koff = hsel * 8;
      afr[mt].h[0] = *(const v8bf*)&lA[cur][row * LDSTR + koff];
      afr[mt].h[1] = *(const v8bf*)&lA[cur][row * LDSTR + koff + 16];
    }
    // ---- B fragments: lane = N col, lane-half picks K {0..7,16..23}/{8..15,24..31}
#pragma unroll
    for (int nt = 0; nt < 4; ++nt) {
      const int col = wn * 64 + nt * 16 + halfl;
      const int kb  = hsel * 8;
      ABf bfr;
      bfr.h[0] = *(const v8bf*)&lB[cur][col * LDSTR + kb];
      bfr.h[1] = *(const v8bf*)&lB[cur][col * LDSTR + kb + 16];
      acc[0][nt] = wmma_bf16(afr[0], bfr, acc[0][nt]);
      acc[1][nt] = wmma_bf16(afr[1], bfr, acc[1][nt]);
    }
    __syncthreads();
    cur ^= 1;
  }

  // ---- epilogue: bias add + store ----
#pragma unroll
  for (int mt = 0; mt < 2; ++mt) {
#pragma unroll
    for (int nt = 0; nt < 4; ++nt) {
      const int col = n0 + wn * 64 + nt * 16 + halfl;
      const float bv = bias[col];
      const int mbase = m0 + wm * 32 + mt * 16 + hsel * 8;
#pragma unroll
      for (int r = 0; r < 8; ++r) {
        const float v = acc[mt][nt][r] + bv;
        const int m = mbase + r;
        if constexpr (OUT_F32) {
          ((float*)out)[(size_t)m * N + col] = v;
        } else {
          const int hh = col >> 7, d = col & 127;   // head-major [h][s][d]
          ((__bf16*)out)[((size_t)hh * M + m) * 128 + d] = (__bf16)v;
        }
      }
    }
  }
}

// ---------------------------------------------------------------------------
// Flash attention per (head, 128-query block).  8 waves; wave w owns score
// rows 16w..16w+15 across all 128 key cols (8 C-tiles) -> softmax reductions
// stay inside 16-lane halves.  K tiles are TDM-DMA'd (ping-pong, next block
// in flight while current computes); P is rewritten into the retired K
// buffer (same-wave DS ordering makes its A-fragment re-read safe).
// ---------------------------------------------------------------------------
#define DPAD 136   // 128 bf16 + 8 pad = 272 B rows (TDM pad_interval/amount match)

__global__ __launch_bounds__(256)
void gqa_flash_attn(const __bf16* __restrict__ Qh, const __bf16* __restrict__ Kg,
                    const __bf16* __restrict__ Vg, const float* __restrict__ mask,
                    __bf16* __restrict__ Abuf, int S) {
  __shared__ __bf16 lQ[128 * DPAD];
  __shared__ __bf16 lK[2][128 * DPAD];  // K tile ping-pong; retired side reused as P
  __shared__ __bf16 lVT[128 * DPAD];    // V transposed: [d][kcol]

  constexpr int D = 128;
  const int h  = blockIdx.y;
  const int g  = h >> 2;                 // rep = H/G = 4
  const int q0 = blockIdx.x * 128;
  const float scale = 0.08838834764831845f;  // 1/sqrt(128)

  const int tid   = threadIdx.x;
  const int lane  = tid & 31;
  const int wv    = tid >> 5;
  const int halfl = lane & 15;
  const int hsel  = lane >> 4;
  const int ldrow = tid >> 1;
  const int ldseg = tid & 1;

  // ---- prologue: bring in Q block (resident) and first K tile ----
#if GQA_TDM
  if (wv == 0) {
    tdm_load_tile128(Qh + ((size_t)h * S + q0) * D, lds_byte_off(&lQ[0]), 128);
    tdm_load_tile128(Kg + (size_t)g * S * D, lds_byte_off(&lK[0][0]), S);
    __builtin_amdgcn_s_wait_tensorcnt((short)0);
  }
  __syncthreads();
#else
  {
    const __bf16* src = Qh + ((size_t)h * S + q0 + ldrow) * D + ldseg * 64;
#pragma unroll
    for (int j = 0; j < 8; ++j)
      *(v8bf*)&lQ[ldrow * DPAD + ldseg * 64 + j * 8] = *(const v8bf*)(src + j * 8);
  }
#endif

  v8f  Oacc[8] = {};
  float mrun[8], lrun[8];
#pragma unroll
  for (int r = 0; r < 8; ++r) { mrun[r] = -3.0e38f; lrun[r] = 0.0f; }

  const int qrow = wv * 16 + halfl;   // A-fragment row for this wave/lane
  int cur = 0;

  for (int kb = 0; kb < S; kb += 128) {
    // ---- (a) kick off async DMA of next K tile into the retired buffer ----
#if GQA_TDM
    if (wv == 0 && kb + 128 < S)
      tdm_load_tile128(Kg + ((size_t)g * S + kb + 128) * D,
                       lds_byte_off(&lK[cur ^ 1][0]), S);
#else
    {
      const __bf16* ksrc = Kg + ((size_t)g * S + kb + ldrow) * D + ldseg * 64;
#pragma unroll
      for (int j = 0; j < 8; ++j)
        *(v8bf*)&lK[cur][ldrow * DPAD + ldseg * 64 + j * 8] = *(const v8bf*)(ksrc + j * 8);
    }
#endif
    // ---- (b) stage V transposed (consumed only after the mid barrier) ----
    {
      const __bf16* vsrc = Vg + ((size_t)g * S + kb + ldrow) * D + ldseg * 64;
#pragma unroll
      for (int j = 0; j < 64; ++j)
        lVT[(ldseg * 64 + j) * DPAD + ldrow] = vsrc[j];
      if (kb + 128 < S)
        __builtin_prefetch(Vg + ((size_t)g * S + kb + 128 + ldrow) * D, 0, 0);
    }
#if !GQA_TDM
    __syncthreads();   // manual path: K tile must be visible before S-compute
#endif

    // ---- (c) S = Q * K^T over D=128 (4 k-steps of 32) ----
    v8f Sacc[8] = {};
#pragma unroll
    for (int ds = 0; ds < 4; ++ds) {
      ABf aq;
      const int koff = ds * 32 + hsel * 8;
      aq.h[0] = *(const v8bf*)&lQ[qrow * DPAD + koff];
      aq.h[1] = *(const v8bf*)&lQ[qrow * DPAD + koff + 16];
#pragma unroll
      for (int nt = 0; nt < 8; ++nt) {
        const int col = nt * 16 + halfl;
        const int kk  = ds * 32 + hsel * 8;
        ABf bk;
        bk.h[0] = *(const v8bf*)&lK[cur][col * DPAD + kk];
        bk.h[1] = *(const v8bf*)&lK[cur][col * DPAD + kk + 16];
        Sacc[nt] = wmma_bf16(aq, bk, Sacc[nt]);
      }
    }
    __syncthreads();   // (d) S reads of lK[cur] done -> P write safe; V visible

    // ---- (e) online softmax on this wave's 16 rows ----
    float rmax[8];
#pragma unroll
    for (int r = 0; r < 8; ++r) rmax[r] = -3.0e38f;
    const int growbase = q0 + wv * 16 + hsel * 8;     // global query row base
#pragma unroll
    for (int nt = 0; nt < 8; ++nt) {
      const int col = kb + nt * 16 + halfl;
      const float* mrow = mask + (size_t)growbase * S + col;
#pragma unroll
      for (int r = 0; r < 8; ++r) {
        float v = Sacc[nt][r] * scale + mrow[(size_t)r * S] * (-1.0e9f);
        Sacc[nt][r] = v;
        rmax[r] = fmaxf(rmax[r], v);
      }
    }
#pragma unroll
    for (int r = 0; r < 8; ++r) {
      float v = rmax[r];
      for (int off = 1; off < 16; off <<= 1)
        v = fmaxf(v, __shfl_xor(v, off, 32));
      rmax[r] = v;
    }
    float esc[8], rsum[8];
#pragma unroll
    for (int r = 0; r < 8; ++r) {
      const float mnew = fmaxf(mrun[r], rmax[r]);
      esc[r]  = __expf(mrun[r] - mnew);
      mrun[r] = mnew;
      rsum[r] = 0.0f;
    }
#pragma unroll
    for (int nt = 0; nt < 8; ++nt)
#pragma unroll
      for (int r = 0; r < 8; ++r) {
        const float p = __expf(Sacc[nt][r] - mrun[r]);
        Sacc[nt][r] = p;
        rsum[r] += p;
      }
#pragma unroll
    for (int r = 0; r < 8; ++r) {
      float v = rsum[r];
      for (int off = 1; off < 16; off <<= 1)
        v += __shfl_xor(v, off, 32);
      lrun[r] = lrun[r] * esc[r] + v;
    }
#pragma unroll
    for (int nt = 0; nt < 8; ++nt)
#pragma unroll
      for (int r = 0; r < 8; ++r) Oacc[nt][r] *= esc[r];

    // ---- write P (bf16) into lK[cur] rows owned by this wave ----
    {
      const int rbase = wv * 16 + hsel * 8;
#pragma unroll
      for (int nt = 0; nt < 8; ++nt) {
        const int col = nt * 16 + halfl;
#pragma unroll
        for (int r = 0; r < 8; ++r)
          lK[cur][(rbase + r) * DPAD + col] = (__bf16)Sacc[nt][r];
      }
    }

    // ---- O += P * V  (A = own P rows, B = V^T tile) ----
#pragma unroll
    for (int kc = 0; kc < 4; ++kc) {
      ABf ap;
      const int koff = kc * 32 + hsel * 8;
      ap.h[0] = *(const v8bf*)&lK[cur][qrow * DPAD + koff];
      ap.h[1] = *(const v8bf*)&lK[cur][qrow * DPAD + koff + 16];
#pragma unroll
      for (int nt = 0; nt < 8; ++nt) {
        const int drow = nt * 16 + halfl;
        const int cc   = kc * 32 + hsel * 8;
        ABf bv;
        bv.h[0] = *(const v8bf*)&lVT[drow * DPAD + cc];
        bv.h[1] = *(const v8bf*)&lVT[drow * DPAD + cc + 16];
        Oacc[nt] = wmma_bf16(ap, bv, Oacc[nt]);
      }
    }

#if GQA_TDM
    if (wv == 0 && kb + 128 < S)
      __builtin_amdgcn_s_wait_tensorcnt((short)0);   // (f) next tile landed
#endif
    __syncthreads();   // (g) protect lVT restage + retired-buffer DMA target
    cur ^= 1;
  }

  // ---- normalize, stage [s_local][d] into lQ, then coalesced permuted store:
  //      Abuf[h*D*S + d*S + (q0+s)]  (faithful to reference's swapaxes+reshape)
  {
    const int rbase = wv * 16 + hsel * 8;
#pragma unroll
    for (int nt = 0; nt < 8; ++nt) {
      const int dcol = nt * 16 + halfl;
#pragma unroll
      for (int r = 0; r < 8; ++r)
        lQ[(rbase + r) * DPAD + dcol] = (__bf16)(Oacc[nt][r] / lrun[r]);
    }
  }
  __syncthreads();
  {
    const int d     = tid & 127;
    const int halfs = tid >> 7;   // 0/1 -> 64 s-values each
    __bf16* dst = Abuf + (size_t)h * D * S + (size_t)d * S + q0 + halfs * 64;
#pragma unroll
    for (int j = 0; j < 8; ++j) {
      v8bf v;
#pragma unroll
      for (int e = 0; e < 8; ++e) v[e] = lQ[(halfs * 64 + j * 8 + e) * DPAD + d];
      *(v8bf*)(dst + j * 8) = v;
    }
  }
}

// ---------------------------------------------------------------------------
extern "C" void kernel_launch(void* const* d_in, const int* in_sizes, int n_in,
                              void* d_out, int out_size, void* d_ws, size_t ws_size,
                              hipStream_t stream) {
  (void)in_sizes; (void)n_in; (void)out_size; (void)ws_size;
  const float* hidden = (const float*)d_in[0];   // [2048, 4096]
  const float* mask   = (const float*)d_in[1];   // [2048, 2048]
  const float* Wq     = (const float*)d_in[2];
  const float* bq     = (const float*)d_in[3];
  const float* Wk     = (const float*)d_in[4];
  const float* bk     = (const float*)d_in[5];
  const float* Wv     = (const float*)d_in[6];
  const float* bv     = (const float*)d_in[7];
  const float* Wo     = (const float*)d_in[8];
  const float* bo     = (const float*)d_in[9];

  const int S = 2048, HID = 4096, NKV = 1024;

  char* ws = (char*)d_ws;
  __bf16* Qh   = (__bf16*)(ws);                         // [32][2048][128] 16 MiB
  __bf16* Kgp  = (__bf16*)(ws + (size_t)16 * 1048576);  // [ 8][2048][128]  4 MiB
  __bf16* Vgp  = (__bf16*)(ws + (size_t)20 * 1048576);  // [ 8][2048][128]  4 MiB
  __bf16* Abuf = (__bf16*)(ws + (size_t)24 * 1048576);  // [2048][4096]    16 MiB

  dim3 blk(256);
  // Q/K/V projections (fp32 in -> bf16 head-major out)
  gqa_gemm_xwt<false, false><<<dim3(HID / 128, S / 128), blk, 0, stream>>>(
      hidden, Wq, bq, Qh, S, HID, HID);
  gqa_gemm_xwt<false, false><<<dim3(NKV / 128, S / 128), blk, 0, stream>>>(
      hidden, Wk, bk, Kgp, S, NKV, HID);
  gqa_gemm_xwt<false, false><<<dim3(NKV / 128, S / 128), blk, 0, stream>>>(
      hidden, Wv, bv, Vgp, S, NKV, HID);
  // Fused flash attention with GQA group sharing
  gqa_flash_attn<<<dim3(S / 128, 32), blk, 0, stream>>>(Qh, Kgp, Vgp, mask, Abuf, S);
  // Output projection (bf16 in -> fp32 out)
  gqa_gemm_xwt<true, true><<<dim3(HID / 128, S / 128), blk, 0, stream>>>(
      Abuf, Wo, bo, d_out, S, HID, HID);
}
